// GNNClassifier_51866025066884
// MI455X (gfx1250) — matrix operand order repaired and checked by reading, not compiled, not run on hardware
//
#include <hip/hip_runtime.h>
#include <hip/hip_fp16.h>

typedef __attribute__((ext_vector_type(16))) _Float16 v16h;
typedef __attribute__((ext_vector_type(8)))  float    v8f;

#define HEADS     4
#define CH        64
#define NGRAPHS   128
#define NEG_SLOPE 0.2f
#define NOUT      256   // HEADS*CH, GEMM output width for both layers

// ------------------------------------------------- CDNA5 async copy helpers

// Issue one 16-byte global -> LDS async copy (ASYNCcnt path).
__device__ __forceinline__ void async_copy_b128(unsigned lds_byte_off, const void* gsrc) {
    asm volatile("global_load_async_to_lds_b128 %0, %1, off"
                 :: "v"(lds_byte_off), "v"(gsrc)
                 : "memory");
}

__device__ __forceinline__ void wait_async_zero() {
#if __has_builtin(__builtin_amdgcn_s_wait_asynccnt)
    __builtin_amdgcn_s_wait_asynccnt(0);
#else
    asm volatile("s_wait_asynccnt 0" ::: "memory");
#endif
}

// ---------------------------------------------------------------- utilities

__global__ void k_f32_to_f16(const float* __restrict__ in, _Float16* __restrict__ out, int n) {
    int i = blockIdx.x * blockDim.x + threadIdx.x;
    if (i < n) out[i] = (_Float16)in[i];
}

// W is [K, N] fp32 row-major; produce Wt [N, K] fp16 so a B-fragment column is contiguous.
__global__ void k_pack_w_t(const float* __restrict__ W, _Float16* __restrict__ Wt, int K, int N) {
    int i = blockIdx.x * blockDim.x + threadIdx.x;
    if (i < K * N) {
        int k = i / N, n = i % N;
        Wt[(size_t)n * K + k] = (_Float16)W[i];
    }
}

__global__ void k_fill_f32(float* __restrict__ p, float v, int n) {
    int i = blockIdx.x * blockDim.x + threadIdx.x;
    if (i < n) p[i] = v;
}

// ---------------------------------------------------------------- WMMA GEMM
// C[M, 256] = A[M, K] * B[K, 256], A/B fp16, C fp32.  K in {256, 64}, M % 16 == 0.
// Block = 128 threads = 4 waves, one 16-row M-tile per block.
// The 16xK A tile (contiguous rows) is staged once into LDS via
// global_load_async_to_lds_b128; each wave then reads its (identical) A
// fragment from LDS and owns a 64-wide N strip (4 WMMA tiles).
__global__ __launch_bounds__(128) void k_gemm_wmma(
        const _Float16* __restrict__ A, const _Float16* __restrict__ Bt,
        float* __restrict__ C, int M, int K) {
    __shared__ _Float16 sA[16 * 256];                 // 8 KB max (K=256)

    const int wave = threadIdx.x >> 5;
    const int lane = threadIdx.x & 31;
    const int hg   = lane >> 4;      // lane half-group (ISA A/B K-split)
    const int l16  = lane & 15;
    const int m0   = blockIdx.x * 16;

    // ---- stage A tile (16*K halves = 32*K bytes, contiguous) into LDS
    {
        const char* atile = (const char*)(A + (size_t)m0 * K);
        const unsigned lds_base = (unsigned)(uintptr_t)&sA[0];   // low 32b = LDS offset
        const int nchunks = 2 * K;                               // 16B chunks
        for (int c = threadIdx.x; c < nchunks; c += blockDim.x)
            async_copy_b128(lds_base + c * 16, atile + c * 16);
        wait_async_zero();
    }
    __syncthreads();

    v8f acc[4];
    acc[0] = v8f{}; acc[1] = v8f{}; acc[2] = v8f{}; acc[3] = v8f{};

    const _Float16* sap = &sA[l16 * K + 8 * hg];

    for (int k0 = 0; k0 < K; k0 += 32) {
        // A fragment (from LDS): elems 0..7 -> K=k0+8*hg+e ; 8..15 -> K=k0+16+8*hg+(e-8)
        v16h a;
        *(uint4*)&a       = *(const uint4*)(sap + k0);
        *((uint4*)&a + 1) = *(const uint4*)(sap + k0 + 16);

        #pragma unroll
        for (int t = 0; t < 4; ++t) {
            // B fragment: elems 0..7 -> K=k0+8*hg+e ; 8..15 -> K=k0+16+8*hg+(e-8) (Bt[N,K])
            const int col = wave * 64 + 16 * t + l16;
            const _Float16* bp = Bt + (size_t)col * K + k0 + 8 * hg;
            v16h b;
            *(uint4*)&b       = *(const uint4*)bp;
            *((uint4*)&b + 1) = *(const uint4*)(bp + 16);
            acc[t] = __builtin_amdgcn_wmma_f32_16x16x32_f16(
                         false, a, false, b, (short)0, acc[t], false, false);
        }
    }

    // C/D layout: VGPR v -> row m0 + v + 8*hg, col = lane&15 within tile
    #pragma unroll
    for (int t = 0; t < 4; ++t) {
        const int col = wave * 64 + 16 * t + l16;
        float* cp = C + (size_t)(m0 + 8 * hg) * NOUT + col;
        #pragma unroll
        for (int v = 0; v < 8; ++v) {
            if (m0 + 8 * hg + v < M) cp[(size_t)v * NOUT] = acc[t][v];
        }
    }
}

// ---------------------------------------------------------------- attention

// a_s[n,hd] = <h[n,hd,:], att_src[hd,:]> ; a_d likewise. One thread per (n,hd).
__global__ __launch_bounds__(256) void k_att_coef(
        const float* __restrict__ h, const float* __restrict__ att_src,
        const float* __restrict__ att_dst, float* __restrict__ a_s,
        float* __restrict__ a_d, int N) {
    __shared__ float s_src[HEADS * CH];
    __shared__ float s_dst[HEADS * CH];
    for (int i = threadIdx.x; i < HEADS * CH; i += blockDim.x) {
        s_src[i] = att_src[i];
        s_dst[i] = att_dst[i];
    }
    __syncthreads();
    int i = blockIdx.x * blockDim.x + threadIdx.x;     // over N*HEADS
    if (i >= N * HEADS) return;
    int n = i >> 2, hd = i & 3;
    const float* hp = h + (size_t)n * NOUT + hd * CH;
    float s = 0.f, d = 0.f;
    #pragma unroll 4
    for (int c = 0; c < CH; c += 4) {
        float4 hv = *(const float4*)(hp + c);
        float4 av = *(const float4*)(s_src + hd * CH + c);
        float4 bv = *(const float4*)(s_dst + hd * CH + c);
        s += hv.x * av.x + hv.y * av.y + hv.z * av.z + hv.w * av.w;
        d += hv.x * bv.x + hv.y * bv.y + hv.z * bv.z + hv.w * bv.w;
    }
    a_s[i] = s;
    a_d[i] = d;
}

// ---------------------------------------------------------------- edge phase

__device__ __forceinline__ void edge_sd(const int* __restrict__ ei, int E, int e,
                                        int& s, int& d) {
    if (e < E) { s = ei[e]; d = ei[E + e]; }
    else       { s = d = e - E; }                      // self loop
}

__device__ __forceinline__ float leaky(float x) {
    return x > 0.f ? x : NEG_SLOPE * x;
}

// sign-aware float atomic max via integer atomics (works for +/- values, init = -inf)
__device__ __forceinline__ void atomicMaxF32(float* addr, float val) {
    if (val >= 0.f) atomicMax((int*)addr, __float_as_int(val));
    else            atomicMin((unsigned int*)addr, __float_as_uint(val));
}

__global__ __launch_bounds__(256) void k_edge_max(
        const int* __restrict__ ei, int E, int EL,
        const float* __restrict__ a_s, const float* __restrict__ a_d,
        float* __restrict__ maxb) {
    int i = blockIdx.x * blockDim.x + threadIdx.x;     // over EL*HEADS
    if (i >= EL * HEADS) return;
    int e = i >> 2, hd = i & 3;
    int s, d; edge_sd(ei, E, e, s, d);
    float l = leaky(a_s[s * HEADS + hd] + a_d[d * HEADS + hd]);
    atomicMaxF32(maxb + d * HEADS + hd, l);
}

__global__ __launch_bounds__(256) void k_edge_expsum(
        const int* __restrict__ ei, int E, int EL,
        const float* __restrict__ a_s, const float* __restrict__ a_d,
        const float* __restrict__ maxb, float* __restrict__ exb,
        float* __restrict__ sumb) {
    int i = blockIdx.x * blockDim.x + threadIdx.x;     // over EL*HEADS
    if (i >= EL * HEADS) return;
    int e = i >> 2, hd = i & 3;
    int s, d; edge_sd(ei, E, e, s, d);
    float l  = leaky(a_s[s * HEADS + hd] + a_d[d * HEADS + hd]);
    float ex = __expf(l - maxb[d * HEADS + hd]);
    exb[i] = ex;
    atomicAdd(sumb + d * HEADS + hd, ex);
}

// 64 threads per edge: thread t -> head t/16, channels 4*(t%16)..+3 (float4 gather,
// 4 scalar atomic adds into the L2-resident accumulator).
__global__ __launch_bounds__(256) void k_edge_scatter(
        const int* __restrict__ ei, int E, int EL,
        const float* __restrict__ h, const float* __restrict__ exb,
        const float* __restrict__ sumb, float* __restrict__ accum) {
    int gid = blockIdx.x * blockDim.x + threadIdx.x;
    int e = gid >> 6;
    if (e >= EL) return;
    int t  = gid & 63;
    int hd = t >> 4;
    int c0 = (t & 15) * 4;
    int s, d; edge_sd(ei, E, e, s, d);
    float coef = exb[e * HEADS + hd] / sumb[d * HEADS + hd];
    const float* hp = h + (size_t)s * NOUT + hd * CH + c0;
    float4 hv = *(const float4*)hp;
    float* op = accum + (size_t)d * NOUT + hd * CH + c0;
    atomicAdd(op + 0, coef * hv.x);
    atomicAdd(op + 1, coef * hv.y);
    atomicAdd(op + 2, coef * hv.z);
    atomicAdd(op + 3, coef * hv.w);
}

// mean over heads + bias + ELU  -> out [N, 64]
__global__ __launch_bounds__(256) void k_finalize(
        const float* __restrict__ accum, const float* __restrict__ bias,
        float* __restrict__ out, int N) {
    int i = blockIdx.x * blockDim.x + threadIdx.x;     // over N*CH
    if (i >= N * CH) return;
    int n = i >> 6, c = i & 63;
    const float* ap = accum + (size_t)n * NOUT + c;
    float v = 0.25f * (ap[0] + ap[CH] + ap[2 * CH] + ap[3 * CH]) + bias[c];
    out[i] = v > 0.f ? v : (__expf(v) - 1.f);
}

// ---------------------------------------------------------------- pooling

__global__ __launch_bounds__(256) void k_pool_scatter(
        const float* __restrict__ x, const int* __restrict__ batch,
        float* __restrict__ sums, float* __restrict__ cnt, int N) {
    int i = blockIdx.x * blockDim.x + threadIdx.x;     // over N*CH
    if (i >= N * CH) return;
    int n = i >> 6, c = i & 63;
    int g = batch[n];
    atomicAdd(sums + g * CH + c, x[i]);
    if (c == 0) atomicAdd(cnt + g, 1.f);
}

__global__ __launch_bounds__(128) void k_classifier(
        const float* __restrict__ sums, const float* __restrict__ cnt,
        const float* __restrict__ w, const float* __restrict__ b,
        float* __restrict__ out) {
    int g = blockIdx.x * blockDim.x + threadIdx.x;
    if (g >= NGRAPHS) return;
    float inv = 1.f / fmaxf(cnt[g], 1.f);
    float acc = 0.f;
    #pragma unroll 4
    for (int c = 0; c < CH; ++c) acc += sums[g * CH + c] * inv * w[c];
    out[g] = acc + b[0];
}

// ---------------------------------------------------------------- launcher

static inline size_t align256(size_t x) { return (x + 255) & ~(size_t)255; }

extern "C" void kernel_launch(void* const* d_in, const int* in_sizes, int n_in,
                              void* d_out, int out_size, void* d_ws, size_t ws_size,
                              hipStream_t stream) {
    // Inputs (setup_inputs order)
    const float* x        = (const float*)d_in[0];
    const float* W1       = (const float*)d_in[1];
    const float* att_s1   = (const float*)d_in[2];
    const float* att_d1   = (const float*)d_in[3];
    const float* bias1    = (const float*)d_in[4];
    const float* W2       = (const float*)d_in[5];
    const float* att_s2   = (const float*)d_in[6];
    const float* att_d2   = (const float*)d_in[7];
    const float* bias2    = (const float*)d_in[8];
    const float* clf_w    = (const float*)d_in[9];
    const float* clf_b    = (const float*)d_in[10];
    const int*   ei       = (const int*)d_in[11];
    const int*   batch    = (const int*)d_in[12];
    float*       out      = (float*)d_out;

    const int IN = 256;
    const int N  = in_sizes[0] / IN;       // 50000
    const int E  = in_sizes[11] / 2;       // 800000
    const int EL = E + N;                  // with self loops

    // Workspace carve-up
    char* ws = (char*)d_ws;
    size_t off = 0;
    auto carve = [&](size_t bytes) { char* p = ws + off; off += align256(bytes); return p; };

    _Float16* xh    = (_Float16*)carve((size_t)N * IN * sizeof(_Float16));   // fp16 x
    _Float16* w1t   = (_Float16*)carve((size_t)IN * NOUT * sizeof(_Float16));
    _Float16* w2t   = (_Float16*)carve((size_t)CH * NOUT * sizeof(_Float16));
    _Float16* midh  = (_Float16*)carve((size_t)N * CH * sizeof(_Float16));   // fp16 layer-2 input
    float*    hbuf  = (float*)carve((size_t)N * NOUT * sizeof(float));       // GEMM out [N,256]
    float*    accum = (float*)carve((size_t)N * NOUT * sizeof(float));       // scatter target
    float*    a_s   = (float*)carve((size_t)N * HEADS * sizeof(float));
    float*    a_d   = (float*)carve((size_t)N * HEADS * sizeof(float));
    float*    maxb  = (float*)carve((size_t)N * HEADS * sizeof(float));
    float*    sumb  = (float*)carve((size_t)N * HEADS * sizeof(float));
    float*    exb   = (float*)carve((size_t)EL * HEADS * sizeof(float));
    float*    mid1  = (float*)carve((size_t)N * CH * sizeof(float));         // ELU(layer1)
    float*    mid2  = (float*)carve((size_t)N * CH * sizeof(float));         // ELU(layer2)
    float*    psum  = (float*)carve((size_t)NGRAPHS * CH * sizeof(float));
    float*    pcnt  = (float*)carve((size_t)NGRAPHS * sizeof(float));
    (void)ws_size;

    const int B = 256;
    auto blocks = [](long long n, int b) { return (unsigned)((n + b - 1) / b); };

    // ---- weight / input packing
    k_f32_to_f16<<<blocks((long long)N * IN, B), B, 0, stream>>>(x, xh, N * IN);
    k_pack_w_t <<<blocks((long long)IN * NOUT, B), B, 0, stream>>>(W1, w1t, IN, NOUT);
    k_pack_w_t <<<blocks((long long)CH * NOUT, B), B, 0, stream>>>(W2, w2t, CH, NOUT);

    // ---- one GAT layer
    auto gat_layer = [&](const _Float16* Ain, int K, const _Float16* Wt,
                         const float* asv, const float* adv, const float* bias,
                         float* layer_out) {
        k_gemm_wmma<<<(unsigned)((N + 15) / 16), 128, 0, stream>>>(Ain, Wt, hbuf, N, K);
        k_att_coef <<<blocks((long long)N * HEADS, B), B, 0, stream>>>(hbuf, asv, adv, a_s, a_d, N);
        k_fill_f32 <<<blocks((long long)N * HEADS, B), B, 0, stream>>>(maxb, -INFINITY, N * HEADS);
        k_fill_f32 <<<blocks((long long)N * HEADS, B), B, 0, stream>>>(sumb, 0.f, N * HEADS);
        k_fill_f32 <<<blocks((long long)N * NOUT, B), B, 0, stream>>>(accum, 0.f, N * NOUT);
        k_edge_max    <<<blocks((long long)EL * HEADS, B), B, 0, stream>>>(ei, E, EL, a_s, a_d, maxb);
        k_edge_expsum <<<blocks((long long)EL * HEADS, B), B, 0, stream>>>(ei, E, EL, a_s, a_d, maxb, exb, sumb);
        k_edge_scatter<<<blocks((long long)EL * 64, B), B, 0, stream>>>(ei, E, EL, hbuf, exb, sumb, accum);
        k_finalize    <<<blocks((long long)N * CH, B), B, 0, stream>>>(accum, bias, layer_out, N);
    };

    // ---- layer 1 (K=256) -> mid1, layer 2 (K=64) -> mid2
    gat_layer(xh, IN, w1t, att_s1, att_d1, bias1, mid1);
    k_f32_to_f16<<<blocks((long long)N * CH, B), B, 0, stream>>>(mid1, midh, N * CH);
    gat_layer(midh, CH, w2t, att_s2, att_d2, bias2, mid2);

    // ---- global mean pool + classifier
    k_fill_f32<<<blocks(NGRAPHS * CH, B), B, 0, stream>>>(psum, 0.f, NGRAPHS * CH);
    k_fill_f32<<<1, B, 0, stream>>>(pcnt, 0.f, NGRAPHS);
    k_pool_scatter<<<blocks((long long)N * CH, B), B, 0, stream>>>(mid2, batch, psum, pcnt, N);
    k_classifier<<<1, 128, 0, stream>>>(psum, pcnt, clf_w, clf_b, out);
}
